// CoreRNNFW_65463891525846
// MI455X (gfx1250) — matrix operand order repaired and checked by reading, not compiled, not run on hardware
//
#include <hip/hip_runtime.h>
#include <hip/hip_bf16.h>

#define T_STEPS 64
#define BATCH   128
#define DG      256
#define DH      512
#define DOUT    256
#define S_LOOP  3
#define LAMBDA  0.9f
#define ETA     0.5f
#define LN_EPS  1e-5f
#define HSTRIDE 528   // DH + 16 halves of padding per history row

typedef __attribute__((ext_vector_type(8)))  __bf16 v8bf;
typedef __attribute__((ext_vector_type(16))) __bf16 v16bf;
typedef __attribute__((ext_vector_type(8)))  float  v8f;

// ---- WMMA fragment loaders -------------------------------------------------
// 16-bit A-matrix 16x32 layout (contiguous k order, matches B): lanes 0-15 hold
// row M=lane with K=0..15; lanes 16-31 hold row M=lane-16 with K=16..31.
__device__ __forceinline__ v16bf load_a_tile(const __bf16* base, int ld,
                                             int m0, int k0, int lane) {
  const int row  = m0 + (lane & 15);
  const int koff = (lane & 16) ? 16 : 0;
  const __bf16* p = base + row * ld + k0 + koff;
  v8bf lo = *(const v8bf*)p;          // K = koff+0..7
  v8bf hi = *(const v8bf*)(p + 8);    // K = koff+8..15
  return __builtin_shufflevector(lo, hi, 0,1,2,3,4,5,6,7,8,9,10,11,12,13,14,15);
}

// B-matrix 32x16, all 16 columns broadcast the same vector x:
// lanes 0-15 hold K=0..15, lanes 16-31 hold K=16..31.
__device__ __forceinline__ v16bf load_b_bcast(const __bf16* x, int k0, int lane) {
  return *(const v16bf*)(x + k0 + ((lane & 16) ? 16 : 0));
}

// ---- block-wide dual sum reduction (512 threads = 16 wave32) ---------------
__device__ __forceinline__ void block_reduce2(float a, float b, float* scr,
                                              int lane, int wave,
                                              float& ra, float& rb) {
  #pragma unroll
  for (int off = 16; off > 0; off >>= 1) {
    a += __shfl_xor(a, off, 32);
    b += __shfl_xor(b, off, 32);
  }
  if (lane == 0) { scr[wave] = a; scr[16 + wave] = b; }
  __syncthreads();
  if (wave == 0) {
    float x = (lane < 16) ? scr[lane]      : 0.0f;
    float y = (lane < 16) ? scr[16 + lane] : 0.0f;
    #pragma unroll
    for (int off = 8; off > 0; off >>= 1) {
      x += __shfl_xor(x, off, 32);
      y += __shfl_xor(y, off, 32);
    }
    if (lane == 0) { scr[0] = x; scr[16] = y; }
  }
  __syncthreads();
  ra = scr[0]; rb = scr[16];
  __syncthreads();   // scr is reused by later reductions
}

// ---- fp32 -> bf16 weight conversion ---------------------------------------
__global__ void cvt_bf16_kernel(const float* __restrict__ src,
                                __bf16* __restrict__ dst, int n) {
  int i = blockIdx.x * blockDim.x + threadIdx.x;
  if (i < n) dst[i] = (__bf16)src[i];
}

// ---- main persistent kernel: one workgroup per batch element ---------------
__global__ __launch_bounds__(512, 1)
void fastweight_rnn_kernel(const float* __restrict__ z_seq,
                           const float* __restrict__ clean_vec,
                           const float* __restrict__ b_h,
                           const float* __restrict__ ln_gamma,
                           const float* __restrict__ ln_beta,
                           const float* __restrict__ head_b,
                           const __bf16* __restrict__ Wh,
                           const __bf16* __restrict__ Wg,
                           const __bf16* __restrict__ Whead,
                           float* __restrict__ partial_loss,
                           float* __restrict__ partial_acc) {
  extern __shared__ char smem[];
  float*  hbase  = (float*)smem;            // 512 f32
  float*  scores = hbase + DH;              // 64  f32
  float*  redbuf = scores + 64;             // 32  f32
  float*  pred   = redbuf + 32;             // 256 f32
  __bf16* cat    = (__bf16*)(pred + DOUT);  // 768 bf16 : [h_prev ; z_t]
  __bf16* hcur   = cat + (DH + DG);         // 512 bf16
  __bf16* Hh     = hcur + DH;               // 64 * HSTRIDE bf16 history

  const int b    = blockIdx.x;
  const int tid  = threadIdx.x;   // 0..511
  const int lane = tid & 31;
  const int wave = tid >> 5;      // 0..15

  cat[tid] = (__bf16)0.0f;        // h0 = 0
  __syncthreads();

  // each wave owns two 16-row output tiles of h_base: rows [m0,m0+16) and
  // [m1,m1+16) with m1 = m0 + 256  (32 tiles / 16 waves)
  const int m0 = wave * 16;
  const int m1 = m0 + (DH / 2);

  for (int t = 0; t < T_STEPS; ++t) {
    // stage z_t[b] into the tail of the concatenated input vector
    if (tid < DG)
      cat[DH + tid] = (__bf16)z_seq[((size_t)t * BATCH + b) * DG + tid];
    __syncthreads();

    // ---- h_base = [Wh | Wg] @ [h ; z_t] + b_h   (rows=512, K=768) ----
    // two independent accumulator chains per wave; one LDS B-load feeds two
    // WMMAs -> more independent work between waitcnts, half the DS traffic.
    {
      v8f acc0 = {}, acc1 = {};
      #pragma unroll 2
      for (int kt = 0; kt < DH / 32; ++kt) {           // Wh part, K=0..511
        const int k0 = kt * 32;
        v16bf bb = load_b_bcast(cat, k0, lane);
        v16bf a0 = load_a_tile(Wh, DH, m0, k0, lane);
        v16bf a1 = load_a_tile(Wh, DH, m1, k0, lane);
        __builtin_prefetch(Wh + (m0 + (lane & 15)) * DH + k0 + 32, 0, 1);
        acc0 = __builtin_amdgcn_wmma_f32_16x16x32_bf16(
                   false, a0, false, bb, (short)0, acc0, false, false);
        acc1 = __builtin_amdgcn_wmma_f32_16x16x32_bf16(
                   false, a1, false, bb, (short)0, acc1, false, false);
      }
      #pragma unroll 2
      for (int kt = 0; kt < DG / 32; ++kt) {           // Wg part, K=512..767
        const int k0 = kt * 32;
        v16bf bb = load_b_bcast(cat, DH + k0, lane);
        v16bf a0 = load_a_tile(Wg, DG, m0, k0, lane);
        v16bf a1 = load_a_tile(Wg, DG, m1, k0, lane);
        acc0 = __builtin_amdgcn_wmma_f32_16x16x32_bf16(
                   false, a0, false, bb, (short)0, acc0, false, false);
        acc1 = __builtin_amdgcn_wmma_f32_16x16x32_bf16(
                   false, a1, false, bb, (short)0, acc1, false, false);
      }
      // all 16 result columns identical: lane 0 holds rows +0..7,
      // lane 16 holds rows +8..15
      if (lane == 0 || lane == 16) {
        const int r0 = m0 + (lane ? 8 : 0);
        const int r1 = m1 + (lane ? 8 : 0);
        #pragma unroll
        for (int r = 0; r < 8; ++r) {
          float v0 = acc0[r] + b_h[r0 + r];
          float v1 = acc1[r] + b_h[r1 + r];
          hbase[r0 + r] = v0;
          hbase[r1 + r] = v1;
          hcur[r0 + r]  = (__bf16)fmaxf(v0, 0.0f);
          hcur[r1 + r]  = (__bf16)fmaxf(v1, 0.0f);
        }
      }
    }
    __syncthreads();

    // ---- inner S-loop: h = relu(LN(h_base + A h)),
    //      A h = eta * sum_s lambda^(t-1-s) * H[s] * (H[s] . h) ----
    const int nhist = t;
    for (int s = 0; s < S_LOOP; ++s) {
      if (nhist > 0) {
        const int ntile = (nhist + 15) >> 4;        // <= 4 history row tiles
        if (wave < ntile) {
          const int hm0 = wave * 16;
          v8f acc = {};
          #pragma unroll 4
          for (int kt = 0; kt < DH / 32; ++kt) {
            v16bf a  = load_a_tile(Hh, HSTRIDE, hm0, kt * 32, lane);
            v16bf bb = load_b_bcast(hcur, kt * 32, lane);
            acc = __builtin_amdgcn_wmma_f32_16x16x32_bf16(
                      false, a, false, bb, (short)0, acc, false, false);
          }
          if (lane == 0 || lane == 16) {
            const int base = hm0 + (lane ? 8 : 0);
            #pragma unroll
            for (int r = 0; r < 8; ++r) {
              const int si = base + r;
              if (si < nhist)
                scores[si] = acc[r] * ETA * __powf(LAMBDA, (float)(t - 1 - si));
            }
          }
        }
      }
      __syncthreads();

      // v = h_base + sum_s scores[s] * H[s][tid]   (one element per thread)
      float vi = hbase[tid];
      for (int r = 0; r < nhist; ++r)
        vi = fmaf(scores[r], (float)Hh[r * HSTRIDE + tid], vi);

      // LayerNorm over 512 elements (biased variance) + relu
      float s1, s2;
      block_reduce2(vi, vi * vi, redbuf, lane, wave, s1, s2);
      const float mu  = s1 * (1.0f / DH);
      const float var = s2 * (1.0f / DH) - mu * mu;
      const float hn  = (vi - mu) * rsqrtf(var + LN_EPS) * ln_gamma[tid] + ln_beta[tid];
      hcur[tid] = (__bf16)fmaxf(hn, 0.0f);
      __syncthreads();
    }

    // Hebbian update == append h to the (undecayed) history; skip on query step
    if (t < T_STEPS - 1) Hh[t * HSTRIDE + tid] = hcur[tid];
    cat[tid] = hcur[tid];       // h_prev for next timestep
    __syncthreads();
  }

  // ---- head readout: pred = Whead @ h + head_b  (rows=256, K=512) ----
  {
    const int hm0 = wave * 16;   // 16 waves x 16 rows = 256 rows
    v8f acc = {};
    #pragma unroll 4
    for (int kt = 0; kt < DH / 32; ++kt) {
      v16bf a  = load_a_tile(Whead, DH, hm0, kt * 32, lane);
      v16bf bb = load_b_bcast(hcur, kt * 32, lane);
      acc = __builtin_amdgcn_wmma_f32_16x16x32_bf16(
                false, a, false, bb, (short)0, acc, false, false);
    }
    if (lane == 0 || lane == 16) {
      const int base = hm0 + (lane ? 8 : 0);
      #pragma unroll
      for (int r = 0; r < 8; ++r)
        pred[base + r] = acc[r] + head_b[base + r];
    }
  }
  __syncthreads();

  // ---- normalized MSE + cosine accuracy (per-batch partials) ----
  float p = (tid < DOUT) ? pred[tid] : 0.0f;
  float c = (tid < DOUT) ? clean_vec[(size_t)b * DOUT + tid] : 0.0f;
  float np2, nc2;
  block_reduce2(p * p, c * c, redbuf, lane, wave, np2, nc2);
  const float pn = p / (sqrtf(np2) + 1e-6f);
  const float cn = c / (sqrtf(nc2) + 1e-6f);
  const float df = pn - cn;
  float ssd, dot;
  block_reduce2(df * df, pn * cn, redbuf, lane, wave, ssd, dot);
  if (tid == 0) {
    partial_loss[b] = ssd * (1.0f / ((float)BATCH * (float)DOUT));
    partial_acc[b]  = dot * (1.0f / (float)BATCH);
  }
}

// ---- deterministic final reduction over the 128 per-batch partials ---------
__global__ void finalize_kernel(const float* __restrict__ pl,
                                const float* __restrict__ pa,
                                float* __restrict__ out) {
  __shared__ float s[2 * BATCH];
  const int tid = threadIdx.x;      // 128 threads
  s[tid]         = pl[tid];
  s[BATCH + tid] = pa[tid];
  __syncthreads();
  for (int off = BATCH / 2; off > 0; off >>= 1) {
    if (tid < off) {
      s[tid]         += s[tid + off];
      s[BATCH + tid] += s[BATCH + tid + off];
    }
    __syncthreads();
  }
  if (tid == 0) { out[0] = s[0]; out[1] = s[BATCH]; }
}

extern "C" void kernel_launch(void* const* d_in, const int* in_sizes, int n_in,
                              void* d_out, int out_size, void* d_ws, size_t ws_size,
                              hipStream_t stream) {
  (void)in_sizes; (void)n_in; (void)out_size; (void)ws_size;
  const float* z_seq  = (const float*)d_in[0];
  const float* clean  = (const float*)d_in[1];
  const float* W_h    = (const float*)d_in[2];
  const float* W_g    = (const float*)d_in[3];
  const float* b_h    = (const float*)d_in[4];
  const float* ln_g   = (const float*)d_in[5];
  const float* ln_b   = (const float*)d_in[6];
  const float* head_W = (const float*)d_in[7];
  const float* head_b = (const float*)d_in[8];

  // workspace layout: bf16 weights + per-batch partials
  __bf16* Whb = (__bf16*)d_ws;
  __bf16* Wgb = Whb + DH * DH;
  __bf16* Whd = Wgb + DH * DG;
  float*  pl  = (float*)(Whd + DOUT * DH);
  float*  pa  = pl + BATCH;

  cvt_bf16_kernel<<<(DH * DH   + 255) / 256, 256, 0, stream>>>(W_h,    Whb, DH * DH);
  cvt_bf16_kernel<<<(DH * DG   + 255) / 256, 256, 0, stream>>>(W_g,    Wgb, DH * DG);
  cvt_bf16_kernel<<<(DOUT * DH + 255) / 256, 256, 0, stream>>>(head_W, Whd, DOUT * DH);

  const size_t smem = (size_t)(DH + 64 + 32 + DOUT) * sizeof(float)
                    + (size_t)(DH + DG + DH) * sizeof(__bf16)
                    + (size_t)64 * HSTRIDE * sizeof(__bf16);   // ~73.6 KB
  hipFuncSetAttribute((const void*)fastweight_rnn_kernel,
                      hipFuncAttributeMaxDynamicSharedMemorySize, (int)smem);

  fastweight_rnn_kernel<<<BATCH, 512, smem, stream>>>(
      z_seq, clean, b_h, ln_g, ln_b, head_b, Whb, Wgb, Whd, pl, pa);

  finalize_kernel<<<1, BATCH, 0, stream>>>(pl, pa, (float*)d_out);
}
